// AttentionBlock_78718160601240
// MI455X (gfx1250) — matrix operand order repaired and checked by reading, not compiled, not run on hardware
//
#include <hip/hip_runtime.h>

// ---------------------------------------------------------------------------
// AttentionBlock for MI455X (gfx1250, wave32, WMMA).
// Roofline: ~34 GFLOP total, ~60 MB HBM traffic -> matrix-op bound.
// Uses v_wmma_f32_16x16x32_bf16 (the CDNA5 dense bf16 path) for all GEMMs
// with f32 accumulation; softmax/groupnorm stats in f32.
// ---------------------------------------------------------------------------

typedef __bf16 bf16;
typedef __attribute__((ext_vector_type(16))) bf16  v16bf;
typedef __attribute__((ext_vector_type(8)))  bf16  v8bf;
typedef __attribute__((ext_vector_type(8)))  float v8f;

namespace {
constexpr int kB = 8, kC = 512, kHW = 1024, kNH = 8, kDH = 64;
constexpr int kGroups = 32, kCPG = kC / kGroups;   // 16 channels / group
constexpr int kC3 = 3 * kC;                        // 1536
constexpr float kEps = 1e-5f;
constexpr float kScale = 0.125f;                   // DH^-0.5
}  // namespace

// ---- fragment loaders ------------------------------------------------------
__device__ __forceinline__ v16bf load2(const bf16* p0, const bf16* p1) {
  v8bf lo = *(const v8bf*)p0;
  v8bf hi = *(const v8bf*)p1;
  return __builtin_shufflevector(lo, hi, 0, 1, 2, 3, 4, 5, 6, 7,
                                 8, 9, 10, 11, 12, 13, 14, 15);
}
// A fragment (16x32, row-major source row `pr` = &M[m][0]):
// lane m = l&15; halves {k0+base..+7, k0+16+base..+7}, base = (l&16)?8:0.
__device__ __forceinline__ v16bf load_afrag(const bf16* pr, int k0, int base) {
  return load2(pr + k0 + base, pr + k0 + 16 + base);
}
// B fragment (32x16, source stored transposed: `pn` = &T[n][0], contiguous k):
// lane n = l&15; halves {k0+base..+7, k0+16+base..+7}, base = (l&16)?8:0.
__device__ __forceinline__ v16bf load_bfrag(const bf16* pn, int k0, int l16) {
  return load2(pn + k0 + (l16 >> 1), pn + k0 + 16 + (l16 >> 1));
}
__device__ __forceinline__ v8f wmma_bf16(v16bf a, v16bf b, v8f c) {
  return __builtin_amdgcn_wmma_f32_16x16x32_bf16(false, a, false, b,
                                                 (short)0, c, false, false);
}

// ---- 1) weight f32 -> bf16 -------------------------------------------------
__global__ __launch_bounds__(256) void cvt_weights(const float* __restrict__ qkvw,
                                                   const float* __restrict__ projw,
                                                   bf16* __restrict__ wq,
                                                   bf16* __restrict__ wp) {
  int i = blockIdx.x * 256 + threadIdx.x;
  if (i < kC3 * kC) wq[i] = (bf16)qkvw[i];
  if (i < kC * kC)  wp[i] = (bf16)projw[i];
}

// ---- 2) GroupNorm -> bf16 transposed [B, HW, C] ----------------------------
__global__ __launch_bounds__(256) void gnorm(const float* __restrict__ x,
                                             const float* __restrict__ gamma,
                                             const float* __restrict__ beta,
                                             bf16* __restrict__ hn) {
  const int b = blockIdx.x / kGroups, g = blockIdx.x % kGroups;
  const int tid = threadIdx.x;
  const float* xg = x + ((size_t)b * kC + (size_t)g * kCPG) * kHW;
  const int n = kCPG * kHW;  // 16384
  float s = 0.f, ss = 0.f;
  for (int i = tid; i < n; i += 256) {
    float v = xg[i];
    s += v; ss += v * v;
  }
  __shared__ float r0[256], r1[256];
  r0[tid] = s; r1[tid] = ss;
  __syncthreads();
  for (int off = 128; off > 0; off >>= 1) {
    if (tid < off) { r0[tid] += r0[tid + off]; r1[tid] += r1[tid + off]; }
    __syncthreads();
  }
  const float mean = r0[0] * (1.0f / n);
  const float var  = r1[0] * (1.0f / n) - mean * mean;
  const float rstd = rsqrtf(var + kEps);
  for (int i = tid; i < n; i += 256) {
    int c = g * kCPG + (i >> 10);   // i / kHW
    int hw = i & (kHW - 1);
    float v = (xg[i] - mean) * rstd * gamma[c] + beta[c];
    hn[((size_t)b * kHW + hw) * kC + c] = (bf16)v;
  }
}

// ---- 3) QKV GEMM: qkv[o,hw] = Wq[o,:] . hn[hw,:]  (per batch) --------------
// Wave tile: 16 (o) x 64 (hw), K loop over C in steps of 32.
__global__ __launch_bounds__(128) void qkv_gemm(const bf16* __restrict__ wq,
                                                const bf16* __restrict__ hn,
                                                const float* __restrict__ bias,
                                                bf16* __restrict__ qb,
                                                bf16* __restrict__ kb,
                                                bf16* __restrict__ vt) {
  const int wave = blockIdx.x * 4 + (threadIdx.x >> 5);
  const int lane = threadIdx.x & 31;
  const int lr = lane & 15;
  const int base = (lane & 16) ? 8 : 0;
  const int l16 = (lane & 16) ? 16 : 0;
  // waves = B * (C3/16) * (HW/64)
  const int ng = wave & 15;                   // hw-tile group (HW/64 = 16)
  int tmp = wave >> 4;
  const int mt = tmp % (kC3 / 16);
  const int b  = tmp / (kC3 / 16);
  const int m0 = mt * 16, n0 = ng * 64;

  const bf16* arow = wq + (size_t)(m0 + lr) * kC;
  const bf16* bmat = hn + (size_t)b * kHW * kC;
  v8f acc[4] = {v8f{}, v8f{}, v8f{}, v8f{}};
  for (int k0 = 0; k0 < kC; k0 += 32) {
    v16bf a = load_afrag(arow, k0, base);
#pragma unroll
    for (int t = 0; t < 4; ++t) {
      const bf16* bp = bmat + (size_t)(n0 + t * 16 + lr) * kC;
      v16bf bm = load_bfrag(bp, k0, l16);
      acc[t] = wmma_bf16(a, bm, acc[t]);
    }
  }
  // epilogue: o = m0 + base + j (same qkv/head split for whole wave tile)
#pragma unroll
  for (int t = 0; t < 4; ++t) {
    const int n = n0 + t * 16 + lr;  // hw
#pragma unroll
    for (int j = 0; j < 8; ++j) {
      const int o = m0 + base + j;
      float v = acc[t][j] + bias[o];
      const int s  = o >> 9;          // 0=q 1=k 2=v
      const int w  = o & 511;
      const int nh = w >> 6;
      const int d  = w & 63;
      const size_t head = (size_t)b * kNH + nh;
      if (s == 0)      qb[(head * kHW + n) * kDH + d] = (bf16)(v * kScale);
      else if (s == 1) kb[(head * kHW + n) * kDH + d] = (bf16)v;
      else             vt[(head * kDH + d) * kHW + n] = (bf16)v;
    }
  }
}

// ---- 4) fused flash attention ---------------------------------------------
// Block = 4 waves x 32; each wave owns 16 query rows; block owns 64 rows of
// one (b, head). Streams over 64-key tiles: QK^T (wmma) -> online softmax
// (shfl_xor row reductions) -> P via LDS transpose -> PV (wmma).
__global__ __launch_bounds__(128) void attn(const bf16* __restrict__ qb,
                                            const bf16* __restrict__ kb,
                                            const bf16* __restrict__ vt,
                                            bf16* __restrict__ hb) {
  __shared__ __align__(16) bf16 plds[4][16][72];  // per-wave P tile, padded
  const int lane = threadIdx.x & 31;
  const int wid  = threadIdx.x >> 5;
  const int lr   = lane & 15;
  const int base = (lane & 16) ? 8 : 0;
  const int l16  = (lane & 16) ? 16 : 0;
  const int qt = blockIdx.x & 15;               // HW/64 = 16 q-tiles
  int tmp = blockIdx.x >> 4;
  const int nh = tmp % kNH;
  const int b  = tmp / kNH;
  const int i0 = qt * 64 + wid * 16;
  const size_t head = (size_t)b * kNH + nh;

  const bf16* qrow = qb + (head * kHW + (i0 + lr)) * kDH;
  const v16bf aq0 = load_afrag(qrow, 0, base);
  const v16bf aq1 = load_afrag(qrow, 32, base);
  const bf16* kbp = kb + head * kHW * kDH;
  const bf16* vbp = vt + head * kDH * kHW;

  float rmax[8], rsum[8];
#pragma unroll
  for (int j = 0; j < 8; ++j) { rmax[j] = -3.0e38f; rsum[j] = 0.f; }
  v8f oacc[4] = {v8f{}, v8f{}, v8f{}, v8f{}};

  for (int jt = 0; jt < kHW; jt += 64) {
    // S tile [16 rows x 64 keys]
    v8f s[4] = {v8f{}, v8f{}, v8f{}, v8f{}};
#pragma unroll
    for (int t = 0; t < 4; ++t) {
      const bf16* kp = kbp + (size_t)(jt + t * 16 + lr) * kDH;
      s[t] = wmma_bf16(aq0, load_bfrag(kp, 0, l16), s[t]);
      s[t] = wmma_bf16(aq1, load_bfrag(kp, 32, l16), s[t]);
    }
    // online softmax: row = (base + j), distributed over a 16-lane half-group
    float bm[8];
#pragma unroll
    for (int j = 0; j < 8; ++j)
      bm[j] = fmaxf(fmaxf(s[0][j], s[1][j]), fmaxf(s[2][j], s[3][j]));
#pragma unroll
    for (int off = 1; off < 16; off <<= 1)
#pragma unroll
      for (int j = 0; j < 8; ++j)
        bm[j] = fmaxf(bm[j], __shfl_xor(bm[j], off, 32));
#pragma unroll
    for (int j = 0; j < 8; ++j) {
      float nm = fmaxf(rmax[j], bm[j]);
      float corr = __expf(rmax[j] - nm);
      rmax[j] = nm;
      rsum[j] *= corr;
#pragma unroll
      for (int t = 0; t < 4; ++t) oacc[t][j] *= corr;
    }
    float ls[8];
#pragma unroll
    for (int j = 0; j < 8; ++j) ls[j] = 0.f;
#pragma unroll
    for (int t = 0; t < 4; ++t)
#pragma unroll
      for (int j = 0; j < 8; ++j) {
        float p = __expf(s[t][j] - rmax[j]);
        s[t][j] = p;
        ls[j] += p;
      }
#pragma unroll
    for (int off = 1; off < 16; off <<= 1)
#pragma unroll
      for (int j = 0; j < 8; ++j) ls[j] += __shfl_xor(ls[j], off, 32);
#pragma unroll
    for (int j = 0; j < 8; ++j) rsum[j] += ls[j];

    // transpose P (C layout -> row-major) through LDS, then PV
    __syncthreads();
#pragma unroll
    for (int t = 0; t < 4; ++t)
#pragma unroll
      for (int j = 0; j < 8; ++j)
        plds[wid][base + j][t * 16 + lr] = (bf16)s[t][j];
    __syncthreads();
    const bf16* pr = &plds[wid][lr][0];
    const v16bf ap0 = load_afrag(pr, 0, base);
    const v16bf ap1 = load_afrag(pr, 32, base);
#pragma unroll
    for (int t = 0; t < 4; ++t) {
      const bf16* vp = vbp + (size_t)(t * 16 + lr) * kHW;  // lane col n = d
      oacc[t] = wmma_bf16(ap0, load_bfrag(vp, jt, l16), oacc[t]);
      oacc[t] = wmma_bf16(ap1, load_bfrag(vp, jt + 32, l16), oacc[t]);
    }
  }

  // finalize & scatter into the reference's reshape: flat [NH,HW,DH] viewed
  // as [C, HW]:  c = nh*64 + (i>>4), hw = (i&15)*64 + d. Store [hw][c] bf16.
#pragma unroll
  for (int j = 0; j < 8; ++j) {
    const float rs = 1.0f / rsum[j];
    const int i = i0 + base + j;
    const int c = nh * 64 + (i >> 4);
    const int hwb = (i & 15) * 64;
#pragma unroll
    for (int t = 0; t < 4; ++t) {
      const int d = t * 16 + lr;
      hb[((size_t)b * kHW + (hwb + d)) * kC + c] = (bf16)(oacc[t][j] * rs);
    }
  }
}

// ---- 5) proj GEMM + bias + residual ---------------------------------------
__global__ __launch_bounds__(128) void proj_gemm(const bf16* __restrict__ wp,
                                                 const bf16* __restrict__ hb,
                                                 const float* __restrict__ bias,
                                                 const float* __restrict__ x,
                                                 float* __restrict__ out) {
  const int wave = blockIdx.x * 4 + (threadIdx.x >> 5);
  const int lane = threadIdx.x & 31;
  const int lr = lane & 15;
  const int base = (lane & 16) ? 8 : 0;
  const int l16 = (lane & 16) ? 16 : 0;
  // waves = B * (C/16) * (HW/64)
  const int ng = wave & 15;
  int tmp = wave >> 4;
  const int mt = tmp % (kC / 16);
  const int b  = tmp / (kC / 16);
  const int m0 = mt * 16, n0 = ng * 64;

  const bf16* arow = wp + (size_t)(m0 + lr) * kC;
  const bf16* bmat = hb + (size_t)b * kHW * kC;
  v8f acc[4] = {v8f{}, v8f{}, v8f{}, v8f{}};
  for (int k0 = 0; k0 < kC; k0 += 32) {
    v16bf a = load_afrag(arow, k0, base);
#pragma unroll
    for (int t = 0; t < 4; ++t) {
      const bf16* bp = bmat + (size_t)(n0 + t * 16 + lr) * kC;
      acc[t] = wmma_bf16(a, load_bfrag(bp, k0, l16), acc[t]);
    }
  }
#pragma unroll
  for (int t = 0; t < 4; ++t) {
    const int n = n0 + t * 16 + lr;
#pragma unroll
    for (int j = 0; j < 8; ++j) {
      const int o = m0 + base + j;
      const size_t idx = ((size_t)b * kC + o) * kHW + n;
      out[idx] = x[idx] + bias[o] + acc[t][j];
    }
  }
}

// ---------------------------------------------------------------------------
extern "C" void kernel_launch(void* const* d_in, const int* in_sizes, int n_in,
                              void* d_out, int out_size, void* d_ws,
                              size_t ws_size, hipStream_t stream) {
  (void)in_sizes; (void)n_in; (void)out_size; (void)ws_size;
  const float* x     = (const float*)d_in[0];
  const float* gamma = (const float*)d_in[1];
  const float* beta  = (const float*)d_in[2];
  const float* qkvw  = (const float*)d_in[3];
  const float* qkvb  = (const float*)d_in[4];
  const float* projw = (const float*)d_in[5];
  const float* projb = (const float*)d_in[6];
  float* out = (float*)d_out;

  // Workspace carve-up (bf16 elements, ~44 MB total)
  bf16* ws = (bf16*)d_ws;
  size_t off = 0;
  bf16* wq = ws + off; off += (size_t)kC3 * kC;             // 1536x512
  bf16* wp = ws + off; off += (size_t)kC * kC;              // 512x512
  bf16* hn = ws + off; off += (size_t)kB * kHW * kC;        // [B,HW,C]
  bf16* qb = ws + off; off += (size_t)kB * kNH * kHW * kDH; // [B,NH,HW,DH]
  bf16* kb = ws + off; off += (size_t)kB * kNH * kHW * kDH;
  bf16* vt = ws + off; off += (size_t)kB * kNH * kDH * kHW; // [B,NH,DH,HW]
  bf16* hb = ws + off;                                      // [B,HW,C]

  cvt_weights<<<(kC3 * kC + 255) / 256, 256, 0, stream>>>(qkvw, projw, wq, wp);
  gnorm<<<kB * kGroups, 256, 0, stream>>>(x, gamma, beta, hn);
  qkv_gemm<<<(kB * (kC3 / 16) * (kHW / 64)) / 4, 128, 0, stream>>>(
      wq, hn, qkvb, qb, kb, vt);
  attn<<<kB * kNH * (kHW / 64), 128, 0, stream>>>(qb, kb, vt, hb);
  proj_gemm<<<(kB * (kC / 16) * (kHW / 64)) / 4, 128, 0, stream>>>(
      wp, hb, projb, x, out);
}
